// Connection_75411035783724
// MI455X (gfx1250) — matrix operand order repaired and checked by reading, not compiled, not run on hardware
//
#include <hip/hip_runtime.h>
#include <hip/hip_bf16.h>
#include <math.h>

// ---------------- types for WMMA ----------------
typedef __bf16  bf16_t;
typedef float   f32x8   __attribute__((ext_vector_type(8)));
typedef __bf16  bf16x8  __attribute__((ext_vector_type(8)));
typedef __bf16  bf16x16 __attribute__((ext_vector_type(16)));

#define WMMA_BF16(a, b, c) \
    __builtin_amdgcn_wmma_f32_16x16x32_bf16(false, (a), false, (b), (short)0, (c), false, false)

__device__ __forceinline__ bf16_t f2bf(float f) {
    union { float f; unsigned u; } v; v.f = f;
    unsigned r = v.u + 0x7FFFu + ((v.u >> 16) & 1u);   // round-to-nearest-even
    unsigned short h = (unsigned short)(r >> 16);
    return __builtin_bit_cast(bf16_t, h);
}

// A-fragment: lanes 0-15 row M=lane, K halves {half*8..+7, 16+half*8..+7}
__device__ __forceinline__ bf16x16 ldfragA(const bf16_t* p) {
    bf16x8 lo = *(const bf16x8*)p;
    bf16x8 hi = *(const bf16x8*)(p + 16);
    return __builtin_shufflevector(lo, hi, 0,1,2,3,4,5,6,7,8,9,10,11,12,13,14,15);
}
// B-fragment: lane holds col N=lane&15, K halves {half*8..+7, 16+half*8..+7}
__device__ __forceinline__ bf16x16 ldfragB(const bf16_t* p) {
    bf16x8 lo = *(const bf16x8*)p;
    bf16x8 hi = *(const bf16x8*)(p + 16);
    return __builtin_shufflevector(lo, hi, 0,1,2,3,4,5,6,7,8,9,10,11,12,13,14,15);
}

// ---------------- generic GEMM: C[M,N] = A[M,K] * W[N,K]^T (+bias, +gelu) ----------
__global__ __launch_bounds__(256)
void gemm_wmma_bf16(const float* __restrict__ A, const float* __restrict__ W,
                    const float* __restrict__ bias, float* __restrict__ C,
                    int M, int N, int K, int act)
{
    __shared__ __align__(16) bf16_t sA[128][40];
    __shared__ __align__(16) bf16_t sW[128][40];

    const int tid   = threadIdx.x;
    const int lane  = tid & 31;
    const int wid   = tid >> 5;
    const int waveM = (wid >> 2) * 64;
    const int waveN = (wid & 3) * 32;
    const int blockM = blockIdx.y * 128;
    const int blockN = blockIdx.x * 128;
    const int ln   = lane & 15;
    const int half = lane >> 4;

    f32x8 acc[4][2];
    for (int i = 0; i < 4; i++)
        for (int j = 0; j < 2; j++)
            for (int e = 0; e < 8; e++) acc[i][j][e] = 0.0f;

    for (int k0 = 0; k0 < K; k0 += 32) {
#pragma unroll
        for (int it = 0; it < 4; it++) {
            int g   = tid + it * 256;
            int row = g >> 3;
            int col = (g & 7) * 4;
            float4 v = make_float4(0.f, 0.f, 0.f, 0.f);
            int m = blockM + row;
            if (m < M) v = *(const float4*)(A + (size_t)m * K + k0 + col);
            sA[row][col + 0] = f2bf(v.x); sA[row][col + 1] = f2bf(v.y);
            sA[row][col + 2] = f2bf(v.z); sA[row][col + 3] = f2bf(v.w);
        }
#pragma unroll
        for (int it = 0; it < 4; it++) {
            int g   = tid + it * 256;
            int row = g >> 3;
            int col = (g & 7) * 4;
            float4 v = make_float4(0.f, 0.f, 0.f, 0.f);
            int n = blockN + row;
            if (n < N) v = *(const float4*)(W + (size_t)n * K + k0 + col);
            sW[row][col + 0] = f2bf(v.x); sW[row][col + 1] = f2bf(v.y);
            sW[row][col + 2] = f2bf(v.z); sW[row][col + 3] = f2bf(v.w);
        }
        if (k0 + 32 < K) {
            __builtin_prefetch(A + (size_t)(blockM + (tid >> 1)) * K + k0 + 32, 0, 1);
            __builtin_prefetch(W + (size_t)(blockN + (tid >> 1)) * K + k0 + 32, 0, 1);
        }
        __syncthreads();

        bf16x16 bfr[2];
#pragma unroll
        for (int j = 0; j < 2; j++)
            bfr[j] = ldfragB(&sW[waveN + j * 16 + ln][half * 8]);
#pragma unroll
        for (int i = 0; i < 4; i++) {
            bf16x16 afr = ldfragA(&sA[waveM + i * 16 + ln][half * 8]);
#pragma unroll
            for (int j = 0; j < 2; j++)
                acc[i][j] = WMMA_BF16(afr, bfr[j], acc[i][j]);
        }
        __syncthreads();
    }

#pragma unroll
    for (int i = 0; i < 4; i++)
#pragma unroll
        for (int j = 0; j < 2; j++)
#pragma unroll
            for (int e = 0; e < 8; e++) {
                int m = blockM + waveM + i * 16 + half * 8 + e;
                int n = blockN + waveN + j * 16 + ln;
                if (m < M && n < N) {
                    float v = acc[i][j][e];
                    if (bias) v += bias[n];
                    if (act == 2) v = 0.5f * v * (1.0f + erff(v * 0.70710678118654752f));
                    C[(size_t)m * N + n] = v;
                }
            }
}

// ---------------- depthwise causal conv (taps=4) + bias + SiLU ----------------
__global__ void conv_silu_kernel(const float* __restrict__ zxbcdt,
                                 const float* __restrict__ conv_w,
                                 const float* __restrict__ conv_b,
                                 float* __restrict__ xBC, int L)
{
    int idx = blockIdx.x * 256 + threadIdx.x;
    if (idx >= L * 2304) return;
    int c = idx % 2304;
    int l = idx / 2304;
    const float* w = conv_w + c * 4;
    float acc = conv_b[c];
#pragma unroll
    for (int j = 0; j < 4; j++) {
        int ls = l - 3 + j;
        if (ls >= 0) acc += zxbcdt[(size_t)ls * 4384 + 2048 + c] * w[j];
    }
    xBC[idx] = acc / (1.0f + expf(-acc));
}

// ---------------- dt = softplus(raw + dt_bias) ----------------
__global__ void dt_softplus_kernel(const float* __restrict__ zxbcdt,
                                   const float* __restrict__ dt_bias,
                                   float* __restrict__ dt, int L)
{
    int idx = blockIdx.x * 256 + threadIdx.x;
    if (idx >= L * 32) return;
    int h = idx & 31;
    int l = idx >> 5;
    float v = zxbcdt[(size_t)l * 4384 + 4352 + h] + dt_bias[h];
    dt[idx] = (v > 20.0f) ? v : log1pf(expf(v));
}

// ---------------- chunked scan pass 1: inclusive cumsum of dt*A per (chunk,head) ---
__global__ __launch_bounds__(256)
void cumsum_kernel(const float* __restrict__ dtb, const float* __restrict__ A_log,
                   float* __restrict__ cumloc)
{
    const int c = blockIdx.x;       // chunk
    const int head = blockIdx.y;    // head
    const int t = threadIdx.x;      // 0..255
    const float A = -expf(A_log[head]);
    __shared__ float s[256];
    s[t] = dtb[(size_t)(c * 256 + t) * 32 + head] * A;
    __syncthreads();
    for (int off = 1; off < 256; off <<= 1) {
        float add = (t >= off) ? s[t - off] : 0.0f;
        __syncthreads();
        s[t] += add;
        __syncthreads();
    }
    cumloc[((size_t)c * 32 + head) * 256 + t] = s[t];
}

// ---------------- chunked scan pass 2: per-chunk local state (WMMA) ----------------
// S_c[p][n] = sum_t exp(cumQ - cum_t)*dt_t*x_t[p] * B_t[n]   (64x128, K=256)
__global__ __launch_bounds__(256)
void chunk_state_kernel(const float* __restrict__ xBC, const float* __restrict__ dtb,
                        const float* __restrict__ cumloc, float* __restrict__ Sbuf)
{
    const int c = blockIdx.x, head = blockIdx.y;
    const int tid = threadIdx.x, lane = tid & 31, wid = tid >> 5;
    const int ln = lane & 15, half = lane >> 4;
    const int l0 = c * 256;

    __shared__ __align__(16) bf16_t sAx[64][40];    // (dt*x decayed)^T : [p][t]
    __shared__ __align__(16) bf16_t sBt[128][40];   // B^T : [n][t]
    __shared__ float scum[256];

    scum[tid] = cumloc[((size_t)c * 32 + head) * 256 + tid];
    __syncthreads();
    const float cend = scum[255];

    const int waveM = (wid >> 2) * 32;   // 0,32
    const int waveN = (wid & 3) * 32;    // 0..96

    f32x8 acc[2][2];
    for (int i = 0; i < 2; i++)
        for (int j = 0; j < 2; j++)
            for (int e = 0; e < 8; e++) acc[i][j][e] = 0.0f;

    for (int k0 = 0; k0 < 256; k0 += 32) {
#pragma unroll
        for (int it = 0; it < 2; it++) {        // x rows -> transposed, scaled
            int g = tid + it * 256;             // 512 float4 groups, 16/row
            int sl = g >> 4; int p4 = (g & 15) * 4;
            int l = l0 + k0 + sl;
            float w = expf(cend - scum[k0 + sl]) * dtb[(size_t)l * 32 + head];
            float4 v = *(const float4*)(xBC + (size_t)l * 2304 + head * 64 + p4);
            sAx[p4 + 0][sl] = f2bf(v.x * w); sAx[p4 + 1][sl] = f2bf(v.y * w);
            sAx[p4 + 2][sl] = f2bf(v.z * w); sAx[p4 + 3][sl] = f2bf(v.w * w);
        }
#pragma unroll
        for (int it = 0; it < 4; it++) {        // B rows -> transposed
            int g = tid + it * 256;             // 1024 float4 groups, 32/row
            int sl = g >> 5; int n4 = (g & 31) * 4;
            float4 v = *(const float4*)(xBC + (size_t)(l0 + k0 + sl) * 2304 + 2048 + n4);
            sBt[n4 + 0][sl] = f2bf(v.x); sBt[n4 + 1][sl] = f2bf(v.y);
            sBt[n4 + 2][sl] = f2bf(v.z); sBt[n4 + 3][sl] = f2bf(v.w);
        }
        __syncthreads();
        bf16x16 bfr[2];
#pragma unroll
        for (int j = 0; j < 2; j++)
            bfr[j] = ldfragB(&sBt[waveN + j * 16 + ln][half * 8]);
#pragma unroll
        for (int i = 0; i < 2; i++) {
            bf16x16 afr = ldfragA(&sAx[waveM + i * 16 + ln][half * 8]);
#pragma unroll
            for (int j = 0; j < 2; j++)
                acc[i][j] = WMMA_BF16(afr, bfr[j], acc[i][j]);
        }
        __syncthreads();
    }

    const size_t sbase = ((size_t)c * 32 + head) * 8192;
#pragma unroll
    for (int i = 0; i < 2; i++)
#pragma unroll
        for (int j = 0; j < 2; j++)
#pragma unroll
            for (int e = 0; e < 8; e++) {
                int p = waveM + i * 16 + half * 8 + e;
                int n = waveN + j * 16 + ln;
                Sbuf[sbase + (size_t)p * 128 + n] = acc[i][j][e];
            }
}

// ---------------- chunked scan pass 3: inter-chunk combine (elementwise, 32 steps) --
__global__ void combine_kernel(const float* __restrict__ Sbuf,
                               const float* __restrict__ cumloc,
                               float* __restrict__ Rbuf)
{
    int idx = blockIdx.x * 256 + threadIdx.x;   // 32 heads * 8192 state elems
    if (idx >= 32 * 8192) return;
    int head = idx >> 13;
    int pn = idx & 8191;
    float R = 0.0f;
    for (int c = 0; c < 32; c++) {
        size_t base = ((size_t)c * 32 + head) * 8192 + pn;
        Rbuf[base] = R;                               // state at chunk start
        float Dfac = expf(cumloc[((size_t)c * 32 + head) * 256 + 255]);
        R = R * Dfac + Sbuf[base];
    }
}

// ---------------- chunked scan pass 4: intra-chunk + inter-chunk outputs (WMMA) -----
// per block: 64 rows of chunk c, head h.
//   G    = C_rows · B^T            (64x256, K=128), masked by [s<=t]*exp(cum_t-cum_s)
//   Y2   = C_rows · R^T            (64x64,  K=128)
//   Y1   = G̃ · (dt*x)^T            (64x64,  K=r0+64)
//   y    = Y1 + exp(cum_t)*Y2 + D*x
__global__ __launch_bounds__(256)
void chunk_intra_kernel(const float* __restrict__ xBC, const float* __restrict__ dtb,
                        const float* __restrict__ cumloc, const float* __restrict__ Rbuf,
                        const float* __restrict__ Dv, float* __restrict__ y)
{
    const int rt = blockIdx.x;        // row tile 0..3
    const int c  = blockIdx.y;        // chunk 0..31
    const int head = blockIdx.z;      // head 0..31
    const int tid = threadIdx.x, lane = tid & 31, wid = tid >> 5;
    const int ln = lane & 15, half = lane >> 4;
    const int l0 = c * 256;
    const int r0 = rt * 64;

    __shared__ __align__(16) bf16_t sA[64][40];     // 5 KB
    __shared__ __align__(16) bf16_t sB[256][40];    // 20 KB
    __shared__ __align__(16) bf16_t sG[64][264];    // 33.8 KB (masked scores)
    __shared__ float scum[256];                     // 1 KB

    scum[tid] = cumloc[((size_t)c * 32 + head) * 256 + tid];

    // ---- step 1: G = C_rows · B^T (64x256, K=128); 8 waves across N ----
    const int waveN1 = wid * 32;
    f32x8 accG[4][2];
    for (int i = 0; i < 4; i++)
        for (int j = 0; j < 2; j++)
            for (int e = 0; e < 8; e++) accG[i][j][e] = 0.0f;

    for (int k0 = 0; k0 < 128; k0 += 32) {
#pragma unroll
        for (int it = 0; it < 2; it++) {            // C rows: 64x32
            int g = tid + it * 256; int row = g >> 3; int col = (g & 7) * 4;
            float4 v = *(const float4*)(xBC + (size_t)(l0 + r0 + row) * 2304 + 2176 + k0 + col);
            sA[row][col + 0] = f2bf(v.x); sA[row][col + 1] = f2bf(v.y);
            sA[row][col + 2] = f2bf(v.z); sA[row][col + 3] = f2bf(v.w);
        }
#pragma unroll
        for (int it = 0; it < 8; it++) {            // B rows: 256x32
            int g = tid + it * 256; int row = g >> 3; int col = (g & 7) * 4;
            float4 v = *(const float4*)(xBC + (size_t)(l0 + row) * 2304 + 2048 + k0 + col);
            sB[row][col + 0] = f2bf(v.x); sB[row][col + 1] = f2bf(v.y);
            sB[row][col + 2] = f2bf(v.z); sB[row][col + 3] = f2bf(v.w);
        }
        __syncthreads();
        bf16x16 bfr[2];
#pragma unroll
        for (int j = 0; j < 2; j++)
            bfr[j] = ldfragB(&sB[waveN1 + j * 16 + ln][half * 8]);
#pragma unroll
        for (int i = 0; i < 4; i++) {
            bf16x16 afr = ldfragA(&sA[i * 16 + ln][half * 8]);
#pragma unroll
            for (int j = 0; j < 2; j++)
                accG[i][j] = WMMA_BF16(afr, bfr[j], accG[i][j]);
        }
        __syncthreads();
    }

    // ---- step 2: causal mask + decay, requantize to bf16 in sG ----
#pragma unroll
    for (int i = 0; i < 4; i++)
#pragma unroll
        for (int j = 0; j < 2; j++)
#pragma unroll
            for (int e = 0; e < 8; e++) {
                int lr = i * 16 + half * 8 + e;       // local row 0..63
                int s  = waveN1 + j * 16 + ln;        // chunk col 0..255
                int tch = r0 + lr;                    // chunk row
                float v = (s <= tch) ? accG[i][j][e] * expf(scum[tch] - scum[s]) : 0.0f;
                sG[lr][s] = f2bf(v);
            }
    __syncthreads();

    // ---- step 3: Y2 = C_rows · R^T (64x64, K=128); waves 4(M) x 2(N) ----
    const int waveM2 = (wid >> 1) * 16;
    const int waveN2 = (wid & 1) * 32;
    const size_t rbase = ((size_t)c * 32 + head) * 8192;
    f32x8 acc2[2];
    for (int j = 0; j < 2; j++)
        for (int e = 0; e < 8; e++) acc2[j][e] = 0.0f;

    for (int k0 = 0; k0 < 128; k0 += 32) {
#pragma unroll
        for (int it = 0; it < 2; it++) {            // C rows again
            int g = tid + it * 256; int row = g >> 3; int col = (g & 7) * 4;
            float4 v = *(const float4*)(xBC + (size_t)(l0 + r0 + row) * 2304 + 2176 + k0 + col);
            sA[row][col + 0] = f2bf(v.x); sA[row][col + 1] = f2bf(v.y);
            sA[row][col + 2] = f2bf(v.z); sA[row][col + 3] = f2bf(v.w);
        }
#pragma unroll
        for (int it = 0; it < 2; it++) {            // R rows: 64x32 into sB[0..63]
            int g = tid + it * 256; int row = g >> 3; int col = (g & 7) * 4;
            float4 v = *(const float4*)(Rbuf + rbase + (size_t)row * 128 + k0 + col);
            sB[row][col + 0] = f2bf(v.x); sB[row][col + 1] = f2bf(v.y);
            sB[row][col + 2] = f2bf(v.z); sB[row][col + 3] = f2bf(v.w);
        }
        __syncthreads();
        bf16x16 a2 = ldfragA(&sA[waveM2 + ln][half * 8]);
#pragma unroll
        for (int j = 0; j < 2; j++) {
            bf16x16 b2 = ldfragB(&sB[waveN2 + j * 16 + ln][half * 8]);
            acc2[j] = WMMA_BF16(a2, b2, acc2[j]);
        }
        __syncthreads();
    }

    // ---- step 4: Y1 = sG · (dt*x)^T (64x64, K = r0+64 effective) ----
    f32x8 acc3[2];
    for (int j = 0; j < 2; j++)
        for (int e = 0; e < 8; e++) acc3[j][e] = 0.0f;

    const int kmax = r0 + 64;
    for (int k0 = 0; k0 < kmax; k0 += 32) {
#pragma unroll
        for (int it = 0; it < 2; it++) {            // x rows -> transposed, dt-scaled
            int g = tid + it * 256;                 // 512 float4 groups, 16/row
            int sl = g >> 4; int p4 = (g & 15) * 4;
            int l = l0 + k0 + sl;
            float dtv = dtb[(size_t)l * 32 + head];
            float4 v = *(const float4*)(xBC + (size_t)l * 2304 + head * 64 + p4);
            sB[p4 + 0][sl] = f2bf(v.x * dtv); sB[p4 + 1][sl] = f2bf(v.y * dtv);
            sB[p4 + 2][sl] = f2bf(v.z * dtv); sB[p4 + 3][sl] = f2bf(v.w * dtv);
        }
        __syncthreads();
        bf16x16 a3 = ldfragA(&sG[waveM2 + ln][k0 + half * 8]);
#pragma unroll
        for (int j = 0; j < 2; j++) {
            bf16x16 b3 = ldfragB(&sB[waveN2 + j * 16 + ln][half * 8]);
            acc3[j] = WMMA_BF16(a3, b3, acc3[j]);
        }
        __syncthreads();
    }

    // ---- epilogue: y = Y1 + exp(cum_t)*Y2 + D*x ----
    const float Dh = Dv[head];
#pragma unroll
    for (int j = 0; j < 2; j++)
#pragma unroll
        for (int e = 0; e < 8; e++) {
            int lr = waveM2 + half * 8 + e;
            int tch = r0 + lr;
            int l = l0 + tch;
            int p = waveN2 + j * 16 + ln;
            float xv = xBC[(size_t)l * 2304 + head * 64 + p];
            float yv = acc3[j][e] + expf(scum[tch]) * acc2[j][e] + Dh * xv;
            y[(size_t)l * 2048 + head * 64 + p] = yv;
        }
}

// ---------------- y = rmsnorm(y * silu(z)) * norm_w, in place, 1 block/row -------
__global__ __launch_bounds__(256)
void gate_norm_kernel(float* __restrict__ y, const float* __restrict__ zxbcdt,
                      const float* __restrict__ norm_w, int L)
{
    int l = blockIdx.x;
    int t = threadIdx.x;
    float vals[8];
    float ss = 0.0f;
#pragma unroll
    for (int i = 0; i < 8; i++) {
        int c = i * 256 + t;
        float yv = y[(size_t)l * 2048 + c];
        float zv = zxbcdt[(size_t)l * 4384 + c];
        float g  = yv * (zv / (1.0f + expf(-zv)));
        vals[i] = g;
        ss += g * g;
    }
    __shared__ float red[256];
    red[t] = ss;
    __syncthreads();
    for (int s = 128; s > 0; s >>= 1) {
        if (t < s) red[t] += red[t + s];
        __syncthreads();
    }
    float rms = rsqrtf(red[0] / 2048.0f + 1e-5f);
#pragma unroll
    for (int i = 0; i < 8; i++) {
        int c = i * 256 + t;
        y[(size_t)l * 2048 + c] = vals[i] * rms * norm_w[c];
    }
}

// ---------------- gather last tokens: h32[i,:] = outy[i*256,:] ----------------
__global__ void gather_last_kernel(const float* __restrict__ outy, float* __restrict__ h32)
{
    int idx = blockIdx.x * 256 + threadIdx.x;
    if (idx >= 32 * 1024) return;
    int i = idx >> 10;
    int c = idx & 1023;
    h32[idx] = outy[(size_t)(i * 256) * 1024 + c];
}

extern "C" void kernel_launch(void* const* d_in, const int* in_sizes, int n_in,
                              void* d_out, int out_size, void* d_ws, size_t ws_size,
                              hipStream_t stream) {
    const float* x           = (const float*)d_in[0];
    const float* in_proj_w   = (const float*)d_in[1];
    const float* conv_w      = (const float*)d_in[2];
    const float* conv_b      = (const float*)d_in[3];
    const float* dt_bias     = (const float*)d_in[4];
    const float* A_log       = (const float*)d_in[5];
    const float* Dp          = (const float*)d_in[6];
    const float* norm_w      = (const float*)d_in[7];
    const float* mamba_out_w = (const float*)d_in[8];
    const float* mlp_w1      = (const float*)d_in[9];
    const float* mlp_b1      = (const float*)d_in[10];
    const float* mlp_w2      = (const float*)d_in[11];
    const float* mlp_b2      = (const float*)d_in[12];
    float* out = (float*)d_out;

    const int L = 8192;

    float* ws = (float*)d_ws;
    size_t o = 0;
    float* zxbcdt = ws + o; o += (size_t)L * 4384;        // 143.7 MB
    float* xBC    = ws + o; o += (size_t)L * 2304;        //  75.5 MB
    float* dtb    = ws + o; o += (size_t)L * 32;
    float* ybuf   = ws + o; o += (size_t)L * 2048;        //  67 MB
    float* cumloc = ws + o; o += (size_t)32 * 32 * 256;   //   1 MB
    float* Sbuf   = ws + o; o += (size_t)32 * 32 * 8192;  //  33.5 MB
    float* Rbuf   = ws + o; o += (size_t)32 * 32 * 8192;  //  33.5 MB
    float* h32    = ws + o; o += (size_t)32 * 1024;
    float* h1     = ws + o; o += (size_t)32 * 4096;
    float* outy   = Sbuf;   // alias: Sbuf dead after combine, outy born after
    (void)ws_size; (void)in_sizes; (void)n_in; (void)out_size;

    dim3 blk(256);

    // 1) in_proj: zxbcdt = u @ in_proj_w^T   (8192 x 4384)
    gemm_wmma_bf16<<<dim3(35, 64), blk, 0, stream>>>(x, in_proj_w, nullptr, zxbcdt,
                                                     L, 4384, 1024, 0);
    // 2) depthwise conv + silu
    conv_silu_kernel<<<dim3((L * 2304 + 255) / 256), blk, 0, stream>>>(zxbcdt, conv_w, conv_b, xBC, L);
    // 3) dt = softplus
    dt_softplus_kernel<<<dim3((L * 32 + 255) / 256), blk, 0, stream>>>(zxbcdt, dt_bias, dtb, L);
    // 4) chunked SSM scan (WMMA formulation)
    cumsum_kernel<<<dim3(32, 32), blk, 0, stream>>>(dtb, A_log, cumloc);
    chunk_state_kernel<<<dim3(32, 32), blk, 0, stream>>>(xBC, dtb, cumloc, Sbuf);
    combine_kernel<<<dim3((32 * 8192 + 255) / 256), blk, 0, stream>>>(Sbuf, cumloc, Rbuf);
    chunk_intra_kernel<<<dim3(4, 32, 32), blk, 0, stream>>>(xBC, dtb, cumloc, Rbuf, Dp, ybuf);
    // 5) gate + rmsnorm (in place)
    gate_norm_kernel<<<dim3(L), blk, 0, stream>>>(ybuf, zxbcdt, norm_w, L);
    // 6) out-proj: outy = y @ mamba_out_w^T   (8192 x 1024)
    gemm_wmma_bf16<<<dim3(8, 64), blk, 0, stream>>>(ybuf, mamba_out_w, nullptr, outy,
                                                    L, 1024, 2048, 0);
    // 7) gather last tokens (stride 256)
    gather_last_kernel<<<dim3(128), blk, 0, stream>>>(outy, h32);
    // 8) MLP1: gelu(h32 @ w1^T + b1)
    gemm_wmma_bf16<<<dim3(32, 1), blk, 0, stream>>>(h32, mlp_w1, mlp_b1, h1,
                                                    32, 4096, 1024, 2);
    // 9) MLP2: out = h1 @ w2^T + b2
    gemm_wmma_bf16<<<dim3(32, 1), blk, 0, stream>>>(h1, mlp_w2, mlp_b2, out,
                                                    32, 4096, 4096, 0);
}
